// DecoderLayer_19980187861718
// MI455X (gfx1250) — matrix operand order repaired and checked by reading, not compiled, not run on hardware
//
#include <hip/hip_runtime.h>
#include <hip/hip_bf16.h>

// DecoderLayer for MI455X (gfx1250): all GEMMs via v_wmma_f32_16x16x32_f16,
// circular convs folded into K=3*Cin GEMMs, block-cooperative weight tiles
// staged into LDS with global_load_async_to_lds_b128, flash-attention softmax.

#define F_LEN 2048
#define BATCH 4
#define DM    256
#define NH    8
#define DH    32
#define DFF   1024

#define LDS_BYTES_256  (64 * (768 + 8) * 2)    // qkv / ffn1 weight tile
#define LDS_BYTES_1024 (64 * (1024 + 8) * 2)   // ffn2 weight tile

typedef _Float16 v16h __attribute__((ext_vector_type(16)));
typedef _Float16 v8h  __attribute__((ext_vector_type(8)));
typedef float    v8f  __attribute__((ext_vector_type(8)));

union HF16 { v8h h[2]; v16h v; };
static_assert(sizeof(HF16) == 32, "HF16 must be 32 bytes");

// -------- CDNA5 async global->LDS path (guarded; falls back clean)
#if defined(__has_builtin)
#if __has_builtin(__builtin_amdgcn_global_load_async_to_lds_b128)
#define HAVE_ASYNC_LDS 1
#endif
#endif

#if defined(HAVE_ASYNC_LDS)
typedef int v4i_ __attribute__((vector_size(16)));
typedef v4i_ __attribute__((address_space(1)))* gv4ip;
typedef v4i_ __attribute__((address_space(3)))* lv4ip;
__device__ __forceinline__ void async_cp16(const void* g, void* l) {
  __builtin_amdgcn_global_load_async_to_lds_b128((gv4ip)(uintptr_t)g,
                                                 (lv4ip)(uintptr_t)l, 0, 0);
}
__device__ __forceinline__ void async_wait0() {
#if __has_builtin(__builtin_amdgcn_s_wait_asynccnt)
  __builtin_amdgcn_s_wait_asynccnt(0);
#else
  asm volatile("s_wait_asynccnt 0x0" ::: "memory");
#endif
}
#endif

// ---------------------------------------------------------------- pack kernels

__global__ void cast_x_kernel(const float* __restrict__ x, _Float16* __restrict__ xh, int n) {
  int i = blockIdx.x * blockDim.x + threadIdx.x;
  if (i < n) xh[i] = (_Float16)x[i];
}

// WqkvT[n][k]: n = 0..767 (q:0-255, k:256-511, v:512-767), k = tap*256 + ci
__global__ void pack_qkv_w(const float* __restrict__ Wq, const float* __restrict__ Wk,
                           const float* __restrict__ Wv, _Float16* __restrict__ WT) {
  int i = blockIdx.x * blockDim.x + threadIdx.x;
  if (i >= 768 * 768) return;
  int n = i / 768, k = i % 768;
  int tap = k / DM, ci = k % DM;
  int sel = n / DM, co = n % DM;
  const float* W = (sel == 0) ? Wq : ((sel == 1) ? Wk : Wv);
  WT[i] = (_Float16)W[(co * DM + ci) * 3 + tap];
}

// W1T[n][k]: n = co (0..1023), k = tap*256 + ci ; W1 orig [1024][256][3]
__global__ void pack_w1(const float* __restrict__ W1, _Float16* __restrict__ WT) {
  int i = blockIdx.x * blockDim.x + threadIdx.x;
  if (i >= DFF * 768) return;
  int n = i / 768, k = i % 768;
  int tap = k / DM, ci = k % DM;
  WT[i] = (_Float16)W1[(n * DM + ci) * 3 + tap];
}

// W2T[n][k]: n = co (0..255), k = tap*1024 + ci ; W2 orig [256][1024][3]
__global__ void pack_w2(const float* __restrict__ W2, _Float16* __restrict__ WT) {
  int i = blockIdx.x * blockDim.x + threadIdx.x;
  if (i >= DM * 3072) return;
  int n = i / 3072, k = i % 3072;
  int tap = k / DFF, ci = k % DFF;
  WT[i] = (_Float16)W2[(n * DFF + ci) * 3 + tap];
}

// ------------------------------------------------- staged conv-GEMM wave tile
// Block = 8 waves sharing one 64-col N-tile; each wave owns a 16-row M-tile.
// Weights staged per K chunk into LDS (async DMA, padded rows); the 3 circular
// tap row pointers are computed once, the k-loop is tap-major with immediate
// offsets, and A fragments ping-pong (no copy, no WMMA->VALU hazard moves).
template <int CIN>
__device__ __forceinline__ void conv_gemm_lds(const _Float16* __restrict__ S,
                                              const _Float16* __restrict__ WT,
                                              _Float16* __restrict__ ldsW,
                                              int b, int f0, int n0, v8f acc[4]) {
  const int KT = 3 * CIN;
  const int KC = (CIN == DM) ? 768 : 1024;   // staged chunk width (tap-aligned)
  const int KROW = KC + 8;                   // padded LDS row stride (halves)
  const int TPC = (CIN == DM) ? 3 : 1;       // taps per staged chunk
  const int NCH = KC / 8;                    // 16B chunks per LDS row
  int tid = threadIdx.x;
  int lane = tid & 31, lo = lane & 15, hi = lane >> 4;
  int fr = f0 + lo;
  int kh8 = hi * 8;                          // A half-select (ISA 16-bit A 16x32)
  int kk16 = hi * 8;                         // B half-select (ISA B 32x16)

  // circular tap row pointers, computed once per wave tile
  const _Float16* rowp[3];
#pragma unroll
  for (int t = 0; t < 3; ++t) {
    int sf = fr + t - 1;
    sf += (sf < 0) ? F_LEN : 0;
    sf -= (sf >= F_LEN) ? F_LEN : 0;
    rowp[t] = S + ((size_t)b * F_LEN + sf) * CIN + kh8;
  }

  auto do4 = [&](const HF16& a, int k0) {    // 4 B frags from LDS + 4 WMMAs
    HF16 bb[4];
#pragma unroll
    for (int j = 0; j < 4; ++j) {
      const _Float16* bp = ldsW + (size_t)(j * 16 + lo) * KROW + k0 + kk16;
      bb[j].h[0] = *(const v8h*)bp;
      bb[j].h[1] = *(const v8h*)(bp + 16);
    }
#pragma unroll
    for (int j = 0; j < 4; ++j)
      acc[j] = __builtin_amdgcn_wmma_f32_16x16x32_f16(false, a.v, false, bb[j].v,
                                                      (short)0, acc[j], false, false);
  };

#pragma unroll 1
  for (int kc = 0; kc < KT; kc += KC) {
    __syncthreads();                         // previous chunk fully consumed
    // ---- stage WT[n0..n0+63][kc..kc+KC-1] -> LDS, KC/32 16B chunks/thread
#if defined(HAVE_ASYNC_LDS)
#pragma unroll
    for (int i = 0; i < KC / 32; ++i) {
      int c = tid + i * 256;
      int row = c / NCH, cc = c % NCH;
      async_cp16(WT + (size_t)(n0 + row) * KT + kc + cc * 8,
                 ldsW + (size_t)row * KROW + cc * 8);
    }
    async_wait0();
#else
#pragma unroll
    for (int i = 0; i < KC / 32; ++i) {
      int c = tid + i * 256;
      int row = c / NCH, cc = c % NCH;
      *(v8h*)(ldsW + (size_t)row * KROW + cc * 8) =
          *(const v8h*)(WT + (size_t)(n0 + row) * KT + kc + cc * 8);
    }
#endif
    __syncthreads();

    int tap0 = kc / CIN;
#pragma unroll
    for (int ts = 0; ts < TPC; ++ts) {
      const _Float16* arow = rowp[tap0 + ts];
      const int cb = ts * CIN;               // K base of this tap within chunk
      HF16 aA, aB;
      aA.h[0] = *(const v8h*)(arow + 0);     // A frag @ ci=0
      aA.h[1] = *(const v8h*)(arow + 16);
#pragma unroll 1
      for (int ci = 0; ci < CIN; ci += 64) { // ping-pong, no register copies
        aB.h[0] = *(const v8h*)(arow + ci + 32);
        aB.h[1] = *(const v8h*)(arow + ci + 48);
        __builtin_prefetch(arow + ci + 128, 0, 3);
        do4(aA, cb + ci);
        if (ci + 64 < CIN) {
          aA.h[0] = *(const v8h*)(arow + ci + 64);
          aA.h[1] = *(const v8h*)(arow + ci + 80);
        }
        do4(aB, cb + ci + 32);
      }
    }
  }
}

// ----------------------------------------------------------------- QKV GEMM
// M = B*F (8192), N = 768, K = 768. Q/K stored [b][h][f][dh], V transposed
// [b][h][dh][f] so attention B-fragments are contiguous.
__global__ void __launch_bounds__(256) qkv_kernel(
    const _Float16* __restrict__ Xh, const _Float16* __restrict__ WT,
    const float* __restrict__ bq, const float* __restrict__ bk,
    const float* __restrict__ bv, _Float16* __restrict__ Qh,
    _Float16* __restrict__ Kh, _Float16* __restrict__ VT) {
  extern __shared__ __align__(16) char smem[];
  _Float16* ldsW = (_Float16*)smem;
  const int NT = 768 / 64;             // 12 n-tiles
  int nt = blockIdx.x % NT;
  int mg = blockIdx.x / NT;            // group of 8 m-tiles
  int mt = mg * 8 + (threadIdx.x >> 5);
  int b = mt >> 7;
  int f0 = (mt & 127) << 4;
  int n0 = nt * 64;
  v8f acc[4];
#pragma unroll
  for (int j = 0; j < 4; ++j)
#pragma unroll
    for (int i = 0; i < 8; ++i) acc[j][i] = 0.0f;
  conv_gemm_lds<DM>(Xh, WT, ldsW, b, f0, n0, acc);

  int lane = threadIdx.x & 31, lo = lane & 15, rb = (lane >> 4) * 8;
#pragma unroll
  for (int j = 0; j < 4; ++j) {
    int n = n0 + j * 16 + lo;
#pragma unroll
    for (int i = 0; i < 8; ++i) {
      int f = f0 + rb + i;
      float v = acc[j][i];
      if (n < 256) {
        v += bq[n];
        int h = n >> 5, dh = n & 31;
        Qh[(((size_t)b * NH + h) * F_LEN + f) * DH + dh] = (_Float16)v;
      } else if (n < 512) {
        int nn = n - 256;
        v += bk[nn];
        int h = nn >> 5, dh = nn & 31;
        Kh[(((size_t)b * NH + h) * F_LEN + f) * DH + dh] = (_Float16)v;
      } else {
        int nn = n - 512;
        v += bv[nn];
        int h = nn >> 5, dh = nn & 31;
        VT[(((size_t)b * NH + h) * DH + dh) * F_LEN + f] = (_Float16)v;
      }
    }
  }
}

// ------------------------------------------------------------ flash attention
// One wave per (b, h, 16-query tile). dh=32 == one WMMA K step. Streams keys
// in blocks of 32 with online softmax; P transposed C->A layout through LDS.
__global__ void __launch_bounds__(256) attn_kernel(
    const _Float16* __restrict__ Qh, const _Float16* __restrict__ Kh,
    const _Float16* __restrict__ VT, _Float16* __restrict__ Ctxh) {
  __shared__ __align__(16) _Float16 plds[8][16][32];
  int wslot = threadIdx.x >> 5;
  int wave = blockIdx.x * (blockDim.x >> 5) + wslot;
  int qt = wave & 127;
  int bh = wave >> 7;                   // b*8+h
  int f0 = qt << 4;
  int lane = threadIdx.x & 31, lo = lane & 15, hi = lane >> 4;
  int kh8 = hi * 8, kk16 = hi * 8;

  // Q as A-fragment (16x32, K = head dim), pre-scaled by 1/sqrt(n_heads)
  const _Float16* qp = Qh + (((size_t)bh) * F_LEN + f0 + lo) * DH + kh8;
  HF16 qa;
  qa.h[0] = *(const v8h*)qp;
  qa.h[1] = *(const v8h*)(qp + 16);
  const _Float16 qs = (_Float16)0.35355339059f;  // 1/sqrt(8)
#pragma unroll
  for (int t = 0; t < 16; ++t) qa.v[t] *= qs;

  float m[8], l[8];
  v8f acc0, acc1;
#pragma unroll
  for (int i = 0; i < 8; ++i) { m[i] = -1e30f; l[i] = 0.0f; acc0[i] = 0.0f; acc1[i] = 0.0f; }

  const _Float16* kbase = Kh + (size_t)bh * F_LEN * DH;
  const _Float16* vbase = VT + (size_t)bh * DH * F_LEN;

  for (int kp = 0; kp < F_LEN; kp += 32) {
    // S = Q K^T for 32 keys (two 16-col WMMAs)
    const _Float16* kp0 = kbase + (size_t)(kp + lo) * DH + kk16;
    HF16 bk0, bk1;
    bk0.h[0] = *(const v8h*)kp0;
    bk0.h[1] = *(const v8h*)(kp0 + 16);
    const _Float16* kp1 = kp0 + 16 * DH;
    bk1.h[0] = *(const v8h*)kp1;
    bk1.h[1] = *(const v8h*)(kp1 + 16);
    __builtin_prefetch(kp0 + 32 * DH, 0, 3);   // next key block
    v8f s0, s1;
#pragma unroll
    for (int i = 0; i < 8; ++i) { s0[i] = 0.0f; s1[i] = 0.0f; }
    s0 = __builtin_amdgcn_wmma_f32_16x16x32_f16(false, qa.v, false, bk0.v, (short)0, s0, false, false);
    s1 = __builtin_amdgcn_wmma_f32_16x16x32_f16(false, qa.v, false, bk1.v, (short)0, s1, false, false);

    // V fragments issued early so the loads overlap the softmax VALU chain
    const _Float16* vp0 = vbase + (size_t)lo * F_LEN + kp + kk16;
    HF16 bv0, bv1;
    bv0.h[0] = *(const v8h*)vp0;
    bv0.h[1] = *(const v8h*)(vp0 + 16);
    const _Float16* vp1 = vp0 + 16 * F_LEN;
    bv1.h[0] = *(const v8h*)vp1;
    bv1.h[1] = *(const v8h*)(vp1 + 16);

    // online softmax; row r lives in vgpr i across the 16 lanes of a half
#pragma unroll
    for (int i = 0; i < 8; ++i) {
      float rv = fmaxf(s0[i], s1[i]);
#pragma unroll
      for (int mk = 1; mk < 16; mk <<= 1) rv = fmaxf(rv, __shfl_xor(rv, mk, 32));
      float nm = fmaxf(m[i], rv);
      float corr = __expf(m[i] - nm);
      m[i] = nm;
      float p0 = __expf(s0[i] - nm);
      float p1 = __expf(s1[i] - nm);
      float rs = p0 + p1;
#pragma unroll
      for (int mk = 1; mk < 16; mk <<= 1) rs += __shfl_xor(rs, mk, 32);
      l[i] = l[i] * corr + rs;
      acc0[i] *= corr;
      acc1[i] *= corr;
      plds[wslot][hi * 8 + i][lo]      = (_Float16)p0;   // C-layout -> LDS
      plds[wslot][hi * 8 + i][16 + lo] = (_Float16)p1;
    }
    // reload P in A-layout (same-wave DS ops are in-order)
    HF16 pa;
    pa.h[0] = *(const v8h*)&plds[wslot][lo][kh8];
    pa.h[1] = *(const v8h*)&plds[wslot][lo][16 + kh8];
    acc0 = __builtin_amdgcn_wmma_f32_16x16x32_f16(false, pa.v, false, bv0.v, (short)0, acc0, false, false);
    acc1 = __builtin_amdgcn_wmma_f32_16x16x32_f16(false, pa.v, false, bv1.v, (short)0, acc1, false, false);
  }

  int b = bh >> 3, h = bh & 7;
#pragma unroll
  for (int i = 0; i < 8; ++i) {
    float inv = 1.0f / l[i];
    int f = f0 + hi * 8 + i;
    size_t ro = ((size_t)b * F_LEN + f) * DM + h * DH;
    Ctxh[ro + lo]      = (_Float16)(acc0[i] * inv);
    Ctxh[ro + 16 + lo] = (_Float16)(acc1[i] * inv);
  }
}

// --------------------------------------------------------------------- FFN
__global__ void __launch_bounds__(256) ffn1_kernel(
    const _Float16* __restrict__ Ctxh, const _Float16* __restrict__ W1T,
    const float* __restrict__ b1, _Float16* __restrict__ Y1h) {
  extern __shared__ __align__(16) char smem[];
  _Float16* ldsW = (_Float16*)smem;
  const int NT = DFF / 64;              // 16
  int nt = blockIdx.x % NT;
  int mg = blockIdx.x / NT;
  int mt = mg * 8 + (threadIdx.x >> 5);
  int b = mt >> 7, f0 = (mt & 127) << 4, n0 = nt * 64;
  v8f acc[4];
#pragma unroll
  for (int j = 0; j < 4; ++j)
#pragma unroll
    for (int i = 0; i < 8; ++i) acc[j][i] = 0.0f;
  conv_gemm_lds<DM>(Ctxh, W1T, ldsW, b, f0, n0, acc);
  int lane = threadIdx.x & 31, lo = lane & 15, rb = (lane >> 4) * 8;
#pragma unroll
  for (int j = 0; j < 4; ++j) {
    int n = n0 + j * 16 + lo;
    float bias = b1[n];
#pragma unroll
    for (int i = 0; i < 8; ++i) {
      int f = f0 + rb + i;
      float v = fmaxf(acc[j][i] + bias, 0.0f);   // fused ReLU
      Y1h[((size_t)b * F_LEN + f) * DFF + n] = (_Float16)v;
    }
  }
}

__global__ void __launch_bounds__(256) ffn2_kernel(
    const _Float16* __restrict__ Y1h, const _Float16* __restrict__ W2T,
    const float* __restrict__ b2, float* __restrict__ Z) {
  extern __shared__ __align__(16) char smem[];
  _Float16* ldsW = (_Float16*)smem;
  const int NT = DM / 64;               // 4
  int nt = blockIdx.x % NT;
  int mg = blockIdx.x / NT;
  int mt = mg * 8 + (threadIdx.x >> 5);
  int b = mt >> 7, f0 = (mt & 127) << 4, n0 = nt * 64;
  v8f acc[4];
#pragma unroll
  for (int j = 0; j < 4; ++j)
#pragma unroll
    for (int i = 0; i < 8; ++i) acc[j][i] = 0.0f;
  conv_gemm_lds<DFF>(Y1h, W2T, ldsW, b, f0, n0, acc);   // K=3072 -> 3 tap chunks
  int lane = threadIdx.x & 31, lo = lane & 15, rb = (lane >> 4) * 8;
#pragma unroll
  for (int j = 0; j < 4; ++j) {
    int n = n0 + j * 16 + lo;
    float bias = b2[n];
#pragma unroll
    for (int i = 0; i < 8; ++i) {
      int f = f0 + rb + i;
      Z[((size_t)b * F_LEN + f) * DM + n] = acc[j][i] + bias;
    }
  }
}

// ------------------------------------------------------- LayerNorm + residual
__global__ void __launch_bounds__(256) ln_kernel(
    const float* __restrict__ Z, const float* __restrict__ x,
    const float* __restrict__ gamma, const float* __restrict__ beta,
    float* __restrict__ out) {
  int row = blockIdx.x * (blockDim.x >> 5) + (threadIdx.x >> 5);  // (b,f)
  int lane = threadIdx.x & 31;
  const float* z = Z + (size_t)row * DM;
  float v[8];
  float s = 0.0f;
#pragma unroll
  for (int t = 0; t < 8; ++t) { v[t] = z[lane + t * 32]; s += v[t]; }
#pragma unroll
  for (int mk = 1; mk < 32; mk <<= 1) s += __shfl_xor(s, mk, 32);
  float mu = s * (1.0f / DM);
  float q = 0.0f;
#pragma unroll
  for (int t = 0; t < 8; ++t) { float d = v[t] - mu; q += d * d; }
#pragma unroll
  for (int mk = 1; mk < 32; mk <<= 1) q += __shfl_xor(q, mk, 32);
  float rs = rsqrtf(q * (1.0f / DM) + 1e-5f);
  const float* xr = x + (size_t)row * DM;
  float* o = out + (size_t)row * DM;
#pragma unroll
  for (int t = 0; t < 8; ++t) {
    int c = lane + t * 32;
    o[c] = (v[t] - mu) * rs * gamma[c] + beta[c] + xr[c];
  }
}

// ------------------------------------------------------------------- launch
extern "C" void kernel_launch(void* const* d_in, const int* in_sizes, int n_in,
                              void* d_out, int out_size, void* d_ws, size_t ws_size,
                              hipStream_t stream) {
  (void)in_sizes; (void)n_in; (void)out_size; (void)ws_size;
  const float* x  = (const float*)d_in[0];
  const float* Wq = (const float*)d_in[1];
  const float* bq = (const float*)d_in[2];
  const float* Wk = (const float*)d_in[3];
  const float* bk = (const float*)d_in[4];
  const float* Wv = (const float*)d_in[5];
  const float* bv = (const float*)d_in[6];
  const float* W1 = (const float*)d_in[7];
  const float* b1 = (const float*)d_in[8];
  const float* W2 = (const float*)d_in[9];
  const float* b2 = (const float*)d_in[10];
  const float* gamma = (const float*)d_in[11];
  const float* beta  = (const float*)d_in[12];
  float* out = (float*)d_out;

  char* ws = (char*)d_ws;
  auto alloc = [&](size_t bytes) -> char* {
    char* p = ws;
    ws += (bytes + 255) & ~(size_t)255;
    return p;
  };
  _Float16* Xh    = (_Float16*)alloc((size_t)BATCH * F_LEN * DM * 2);
  _Float16* WqkvT = (_Float16*)alloc((size_t)768 * 768 * 2);
  _Float16* W1T   = (_Float16*)alloc((size_t)DFF * 768 * 2);
  _Float16* W2T   = (_Float16*)alloc((size_t)DM * 3072 * 2);
  _Float16* Qh    = (_Float16*)alloc((size_t)BATCH * F_LEN * DM * 2);
  _Float16* Kh    = (_Float16*)alloc((size_t)BATCH * F_LEN * DM * 2);
  _Float16* VT    = (_Float16*)alloc((size_t)BATCH * F_LEN * DM * 2);
  _Float16* Ctxh  = (_Float16*)alloc((size_t)BATCH * F_LEN * DM * 2);
  _Float16* Y1h   = (_Float16*)alloc((size_t)BATCH * F_LEN * DFF * 2);
  float*    Z     = (float*)alloc((size_t)BATCH * F_LEN * DM * 4);

  int nx = BATCH * F_LEN * DM;
  cast_x_kernel<<<(nx + 255) / 256, 256, 0, stream>>>(x, Xh, nx);
  pack_qkv_w<<<(768 * 768 + 255) / 256, 256, 0, stream>>>(Wq, Wk, Wv, WqkvT);
  pack_w1<<<(DFF * 768 + 255) / 256, 256, 0, stream>>>(W1, W1T);
  pack_w2<<<(DM * 3072 + 255) / 256, 256, 0, stream>>>(W2, W2T);

  qkv_kernel<<<64 * 12, 256, LDS_BYTES_256, stream>>>(Xh, WqkvT, bq, bk, bv, Qh, Kh, VT);
  attn_kernel<<<(BATCH * NH * (F_LEN / 16)) / 8, 256, 0, stream>>>(Qh, Kh, VT, Ctxh);
  ffn1_kernel<<<64 * 16, 256, LDS_BYTES_256, stream>>>(Ctxh, W1T, b1, Y1h);
  ffn2_kernel<<<64 * 4, 256, LDS_BYTES_1024, stream>>>(Y1h, W2T, b2, Z);
  ln_kernel<<<(BATCH * F_LEN) / 8, 256, 0, stream>>>(Z, x, gamma, beta, out);
}
